// TextureCoordinateMLP_75746043232979
// MI455X (gfx1250) — compile-verified, matrix operand order checked by reading, not run on hardware
//
#include <hip/hip_runtime.h>
#include <hip/hip_bf16.h>

typedef __attribute__((ext_vector_type(16))) __bf16 v16bf;
typedef __attribute__((ext_vector_type(8)))  __bf16 v8bf;
typedef __attribute__((ext_vector_type(8)))  float  v8f;

#define NCH   8
#define HID   256
#define KPAD  32     // 29 inputs padded to 32
#define NHL   6      // hidden layers

// ---------------------------------------------------------------------------
// Row loader per CDNA5 WMMA VGPR layouts (CDNA5 ISA §7.12.2).
// 16-bit A 16x32 and B 32x16 share the same per-lane pattern:
// lane<16 holds index=lane with K={0..7,16..23}; lane>=16 holds index=lane-16
// with K={8..15,24..31}. rowk points at (row, k0).
// ---------------------------------------------------------------------------
__device__ __forceinline__ v16bf load_row16(const __bf16* rowk, int lane) {
  const int c1 = (lane < 16) ? 0 : 8;
  v8bf lo = *(const v8bf*)(rowk + c1);
  v8bf hi = *(const v8bf*)(rowk + c1 + 16);
  return __builtin_shufflevector(lo, hi, 0,1,2,3,4,5,6,7,8,9,10,11,12,13,14,15);
}

__device__ __forceinline__ v8f wmma_bf16(v16bf a, v16bf b, v8f c) {
  // D = A(16x32 bf16) * B(32x16 bf16) + C(16x16 f32)
  return __builtin_amdgcn_wmma_f32_16x16x32_bf16(
      /*neg_a=*/false, a, /*neg_b=*/false, b,
      /*c_mod=*/(short)0, c, /*reuse_a=*/false, /*reuse_b=*/false);
}

// relu via single v_med3 clamp (avoids fmaxf's canonicalize+max pair)
__device__ __forceinline__ float relu1(float x) {
  return __builtin_amdgcn_fmed3f(x, 0.0f, 3.38953139e38f);
}

// relu two D tiles (neuron rows mt*32..+15 and +16..+31) and pack them into
// the next layer's B operand for K-step mt. Pure intra-lane: D vgpr j of the
// low tile -> element j; of the high tile -> element 8+j.
__device__ __forceinline__ v16bf relu_pack(v8f d0, v8f d1) {
  v16bf r;
#pragma unroll
  for (int e = 0; e < 8; ++e) r[e]     = (__bf16)relu1(d0[e]);
#pragma unroll
  for (int e = 0; e < 8; ++e) r[8 + e] = (__bf16)relu1(d1[e]);
  return r;
}

// One 256->256 hidden layer for TWO point tiles (N=32), fully in registers.
// Each A (weight) tile load feeds two WMMAs -> 1:1 load:wmma instruction
// ratio and half the weight traffic per FLOP. W is [n_out][k_in] bf16.
__device__ __forceinline__ void hidden_layer2(const __bf16* __restrict__ W,
                                              const float* __restrict__ bias,
                                              const v16bf (&B0in)[8], const v16bf (&B1in)[8],
                                              v16bf (&B0out)[8], v16bf (&B1out)[8],
                                              const __bf16* __restrict__ Wnext,
                                              int lane, int mcol, int hiHalf) {
#pragma unroll
  for (int mt = 0; mt < 8; ++mt) {
    __builtin_prefetch(Wnext + (size_t)(mt * 32 + mcol) * HID, 0, 3);
    const v8f bias0 = *(const v8f*)(bias + mt * 32 + hiHalf * 8);       // rows mt*32..+15
    const v8f bias1 = *(const v8f*)(bias + mt * 32 + 16 + hiHalf * 8);  // rows +16..+31
    v8f a00 = bias0, a01 = bias0;           // A-tile0 x {B0, B1}
    v8f a10 = bias1, a11 = bias1;           // A-tile1 x {B0, B1}
    const __bf16* rowA0 = W + (size_t)(mt * 32 + mcol) * HID;
    const __bf16* rowA1 = rowA0 + 16 * HID;
#pragma unroll
    for (int k = 0; k < 8; ++k) {
      v16bf A0 = load_row16(rowA0 + k * 32, lane);
      v16bf A1 = load_row16(rowA1 + k * 32, lane);
      a00 = wmma_bf16(A0, B0in[k], a00);
      a01 = wmma_bf16(A0, B1in[k], a01);
      a10 = wmma_bf16(A1, B0in[k], a10);
      a11 = wmma_bf16(A1, B1in[k], a11);
    }
    B0out[mt] = relu_pack(a00, a10);
    B1out[mt] = relu_pack(a01, a11);
  }
}

// ---------------------------------------------------------------------------
// Prep kernels
// ---------------------------------------------------------------------------
__global__ void zero_counts_kernel(int* counts) {
  if (threadIdx.x < NCH) counts[threadIdx.x] = 0;
}

__global__ void bin_kernel(const int* __restrict__ mlp_idx, int* __restrict__ counts,
                           int* __restrict__ bins, int n) {
  int i = blockIdx.x * blockDim.x + threadIdx.x;
  if (i >= n) return;
  int c = mlp_idx[i];
  int pos = atomicAdd(counts + c, 1);
  bins[c * n + pos] = i;
}

// positional encoding -> bf16, K padded 29->32 with zeros
__global__ void pe_kernel(const float* __restrict__ x, const float* __restrict__ proc,
                          __bf16* __restrict__ Xp, int n) {
  int i = blockIdx.x * blockDim.x + threadIdx.x;
  if (i >= n) return;
  float v0 = x[3 * i + 0], v1 = x[3 * i + 1], v2 = x[3 * i + 2];
  __bf16* row = Xp + (size_t)i * KPAD;
  row[0] = (__bf16)v0; row[1] = (__bf16)v1; row[2] = (__bf16)v2;
  int idx = 3;
#pragma unroll
  for (int d = 1; d <= 4; ++d) {
    float s = 3.14159265358979323846f * (float)(1 << d);
    row[idx + 0] = (__bf16)__sinf(s * v0);
    row[idx + 1] = (__bf16)__sinf(s * v1);
    row[idx + 2] = (__bf16)__sinf(s * v2);
    row[idx + 3] = (__bf16)__cosf(s * v0);
    row[idx + 4] = (__bf16)__cosf(s * v1);
    row[idx + 5] = (__bf16)__cosf(s * v2);
    idx += 6;
  }
  row[27] = (__bf16)proc[0];
  row[28] = (__bf16)proc[1];
  row[29] = (__bf16)0.0f; row[30] = (__bf16)0.0f; row[31] = (__bf16)0.0f;
}

// W0 (8,29,256) f32 -> Wt0 [c][n=256][k=32] bf16, k>=29 zero
__global__ void tr_w0_kernel(const float* __restrict__ W0, __bf16* __restrict__ Wt0) {
  int idx = blockIdx.x * blockDim.x + threadIdx.x;     // NCH*256*32
  if (idx >= NCH * HID * KPAD) return;
  int k = idx & 31, nn = (idx >> 5) & 255, c = idx >> 13;
  float v = (k < 29) ? W0[((size_t)c * 29 + k) * HID + nn] : 0.0f;
  Wt0[idx] = (__bf16)v;
}

// Wh (8,6,256,256) f32 -> Wth [cl][n][k] bf16
__global__ void tr_wh_kernel(const float* __restrict__ Wh, __bf16* __restrict__ Wth) {
  int idx = blockIdx.x * blockDim.x + threadIdx.x;     // NCH*NHL*256*256
  if (idx >= NCH * NHL * HID * HID) return;
  int k = idx & 255, nn = (idx >> 8) & 255, cl = idx >> 16;
  Wth[idx] = (__bf16)Wh[((size_t)cl * HID + k) * HID + nn];
}

// Wl (8,256,2) f32 -> Wtl [c][n=16 padded][k=256] bf16, n>=2 zero
__global__ void tr_wl_kernel(const float* __restrict__ Wl, __bf16* __restrict__ Wtl) {
  int idx = blockIdx.x * blockDim.x + threadIdx.x;     // NCH*16*256
  if (idx >= NCH * 16 * HID) return;
  int k = idx & 255, nn = (idx >> 8) & 15, c = idx >> 12;
  float v = (nn < 2) ? Wl[((size_t)c * HID + k) * 2 + nn] : 0.0f;
  Wtl[idx] = (__bf16)v;
}

// ---------------------------------------------------------------------------
// Main fused MLP kernel. 128 threads = 4 waves, each wave = 32 points (two
// 16-point N-tiles) of one chart. Weights are the A operand (16 out-neurons x
// 32 in-neurons), points are the N dimension of B/C/D. Each weight tile load
// feeds two WMMAs. Activations chain entirely in registers: relu(D-tile pair)
// == next B operand, intra-lane. No LDS. __launch_bounds__(128,1): fat-
// register config (~330 VGPRs live) to avoid accumulator spills.
// ---------------------------------------------------------------------------
__global__ void __launch_bounds__(128, 1)
mlp_kernel(const int* __restrict__ bins, const int* __restrict__ counts,
           const __bf16* __restrict__ Xp,
           const __bf16* __restrict__ Wt0,
           const __bf16* __restrict__ Wth,
           const __bf16* __restrict__ Wtl,
           const float* __restrict__ b0,
           const float* __restrict__ bh,
           const float* __restrict__ bl,
           float* __restrict__ out, int n) {
  const int c = blockIdx.y;
  const int cnt = counts[c];
  const int blkStart = blockIdx.x * 128;
  if (blkStart >= cnt) return;                      // block-uniform exit
  const int wave = threadIdx.x >> 5;
  const int lane = threadIdx.x & 31;
  const int rowBase = blkStart + wave * 32;
  if (rowBase >= cnt) return;                       // wave-uniform exit (EXEC stays full)

  const int mcol = lane & 15;                       // A-row / B-col index per lane
  const int hiHalf = lane >> 4;                     // C/D rows +8 for lanes 16..31
  const int cbase = c * n;

  // gather this wave's 32 points; clamps keep EXEC all-ones through WMMAs
  int r0 = rowBase + mcol;      if (r0 >= cnt) r0 = cnt - 1;
  int r1 = rowBase + 16 + mcol; if (r1 >= cnt) r1 = cnt - 1;
  const int p0 = bins[cbase + r0];
  const int p1 = bins[cbase + r1];

  v16bf B0a[8], B1a[8], B0b[8], B1b[8];             // ping-pong activation banks x2 tiles

  // ------ layer 0: B = gathered PE rows (K=32), A = W0t ------
  {
    B0a[0] = load_row16(Xp + (size_t)p0 * KPAD, lane);
    B1a[0] = load_row16(Xp + (size_t)p1 * KPAD, lane);
    const __bf16* W    = Wt0 + (size_t)c * HID * KPAD;
    const float*  bias = b0 + c * HID;
    const __bf16* Wnext = Wth + (size_t)(c * NHL) * HID * HID;
#pragma unroll
    for (int mt = 0; mt < 8; ++mt) {
      __builtin_prefetch(Wnext + (size_t)(mt * 32 + mcol) * HID, 0, 3);
      const v8f bias0 = *(const v8f*)(bias + mt * 32 + hiHalf * 8);
      const v8f bias1 = *(const v8f*)(bias + mt * 32 + 16 + hiHalf * 8);
      v8f a00 = bias0, a01 = bias0, a10 = bias1, a11 = bias1;
      v16bf A0 = load_row16(W + (size_t)(mt * 32 + mcol) * KPAD, lane);
      v16bf A1 = load_row16(W + (size_t)(mt * 32 + 16 + mcol) * KPAD, lane);
      a00 = wmma_bf16(A0, B0a[0], a00);
      a01 = wmma_bf16(A0, B1a[0], a01);
      a10 = wmma_bf16(A1, B0a[0], a10);
      a11 = wmma_bf16(A1, B1a[0], a11);
      B0b[mt] = relu_pack(a00, a10);
      B1b[mt] = relu_pack(a01, a11);
    }
  }

  // ------ 6 hidden layers, ping-pong b <-> a banks (no register copies) ------
  const __bf16* Wh0 = Wth + (size_t)(c * NHL) * HID * HID;
  const float*  bh0 = bh + c * NHL * HID;
  const __bf16* Wfin = Wtl + (size_t)c * 16 * HID;
  for (int ll = 0; ll < NHL / 2; ++ll) {
    const int l0 = 2 * ll, l1 = 2 * ll + 1;
    const __bf16* Wn1 = (l1 + 1 < NHL) ? Wh0 + (size_t)(l1 + 1) * HID * HID : Wfin;
    hidden_layer2(Wh0 + (size_t)l0 * HID * HID, bh0 + l0 * HID,
                  B0b, B1b, B0a, B1a,
                  Wh0 + (size_t)l1 * HID * HID, lane, mcol, hiHalf);
    hidden_layer2(Wh0 + (size_t)l1 * HID * HID, bh0 + l1 * HID,
                  B0a, B1a, B0b, B1b,
                  Wn1, lane, mcol, hiHalf);
  }

  // ------ final layer: 256 -> 2 (N-rows padded to 16), sigmoid, scatter ------
  {
    v8f acc0 = {}, acc1 = {};
    const float blv0 = (hiHalf == 0) ? bl[c * 2 + 0] : 0.0f;  // row m=0 -> neuron 0
    const float blv1 = (hiHalf == 0) ? bl[c * 2 + 1] : 0.0f;  // row m=1 -> neuron 1
    acc0[0] = blv0; acc0[1] = blv1;
    acc1[0] = blv0; acc1[1] = blv1;
    const __bf16* rowA = Wfin + (size_t)mcol * HID;
#pragma unroll
    for (int k = 0; k < 8; ++k) {
      v16bf A = load_row16(rowA + k * 32, lane);
      acc0 = wmma_bf16(A, B0b[k], acc0);
      acc1 = wmma_bf16(A, B1b[k], acc1);
    }
    // D: lane<16 holds point col=lane, vgpr0/1 = output neurons 0/1
    if (lane < 16) {                                // divergence only after all WMMAs
      const int ra = rowBase + lane;
      if (ra < cnt) {
        const int p = bins[cbase + ra];
        float2 o;
        o.x = 1.0f / (1.0f + __expf(-acc0[0]));
        o.y = 1.0f / (1.0f + __expf(-acc0[1]));
        *(float2*)(out + (size_t)p * 2) = o;        // global_store_b64 per point
      }
      const int rb = rowBase + 16 + lane;
      if (rb < cnt) {
        const int p = bins[cbase + rb];
        float2 o;
        o.x = 1.0f / (1.0f + __expf(-acc1[0]));
        o.y = 1.0f / (1.0f + __expf(-acc1[1]));
        *(float2*)(out + (size_t)p * 2) = o;
      }
    }
  }
}

// ---------------------------------------------------------------------------
extern "C" void kernel_launch(void* const* d_in, const int* in_sizes, int n_in,
                              void* d_out, int out_size, void* d_ws, size_t ws_size,
                              hipStream_t stream) {
  const float* x    = (const float*)d_in[0];
  const float* proc = (const float*)d_in[1];
  const float* W0   = (const float*)d_in[2];
  const float* b0   = (const float*)d_in[3];
  const float* Wh   = (const float*)d_in[4];
  const float* bh   = (const float*)d_in[5];
  const float* Wl   = (const float*)d_in[6];
  const float* bl   = (const float*)d_in[7];
  const int* mlp_idx = (const int*)d_in[8];
  float* out = (float*)d_out;
  const int n = in_sizes[0] / 3;                      // 16384

  // workspace carve-up (all recomputed every call -> deterministic)
  char* ws = (char*)d_ws;
  size_t off = 0;
  __bf16* Xp  = (__bf16*)(ws + off); off += (size_t)n * KPAD * sizeof(__bf16);
  __bf16* Wt0 = (__bf16*)(ws + off); off += (size_t)NCH * HID * KPAD * sizeof(__bf16);
  __bf16* Wth = (__bf16*)(ws + off); off += (size_t)NCH * NHL * HID * HID * sizeof(__bf16);
  __bf16* Wtl = (__bf16*)(ws + off); off += (size_t)NCH * 16 * HID * sizeof(__bf16);
  int* counts = (int*)(ws + off);    off += 256;
  int* bins   = (int*)(ws + off);    off += (size_t)NCH * n * sizeof(int);

  zero_counts_kernel<<<1, 32, 0, stream>>>(counts);
  bin_kernel<<<(n + 255) / 256, 256, 0, stream>>>(mlp_idx, counts, bins, n);
  pe_kernel<<<(n + 255) / 256, 256, 0, stream>>>(x, proc, Xp, n);
  tr_w0_kernel<<<(NCH * HID * KPAD) / 256, 256, 0, stream>>>(W0, Wt0);
  tr_wh_kernel<<<(NCH * NHL * HID * HID) / 256, 256, 0, stream>>>(Wh, Wth);
  tr_wl_kernel<<<(NCH * 16 * HID) / 256, 256, 0, stream>>>(Wl, Wtl);

  dim3 grid(n / 128, NCH, 1);                         // 4 waves * 32 pts per block
  mlp_kernel<<<grid, 128, 0, stream>>>(
      bins, counts, Xp, Wt0, Wth, Wtl, b0, bh, bl, out, n);
}
